// AttentionMechanism_72430328480542
// MI455X (gfx1250) — compile-verified, matrix operand order checked by reading, not run on hardware
//
#include <hip/hip_runtime.h>
#include <hip/hip_bf16.h>
#include <cmath>

typedef __attribute__((ext_vector_type(16))) _Float16 v16h;
typedef __attribute__((ext_vector_type(8)))  _Float16 v8h;
typedef __attribute__((ext_vector_type(8)))  float    v8f;

#define B_     32
#define KLEN_  2048
#define DDIM_  512   // KDIM == QDIM
#define ADIM_  512
#define VDIM_  512
#define NEG_INF_ (-3.4028234663852886e+38f)

// LDS row stride in halves: 40 halves = 80 B = 20 dwords; 16 consecutive rows at
// the same offset hit 16 distinct bank groups (conflict-free b128 frag loads).
#define HPAD_ 40

// Native CDNA5 transcendental tanh (TRANS op, co-executes with WMMA issue).
#if defined(__has_builtin)
#  if __has_builtin(__builtin_amdgcn_tanhf)
#    define TANH_(x) __builtin_amdgcn_tanhf(x)
#  endif
#endif
#ifndef TANH_
#  define TANH_(x) tanhf(x)
#endif

// ---------------------------------------------------------------------------
// q projection: qp[b][a] = sum_d query[b][d] * Wq[a][d]
// ---------------------------------------------------------------------------
__global__ __launch_bounds__(256) void qproj_kernel(
    const float* __restrict__ query, const float* __restrict__ Wq,
    float* __restrict__ qp) {
  __shared__ float sq[DDIM_];
  const int b = blockIdx.x;
  const int t = threadIdx.x;
  for (int d = t; d < DDIM_; d += 256) sq[d] = query[b * DDIM_ + d];
  __syncthreads();
  for (int a = t; a < ADIM_; a += 256) {
    const float* w = Wq + (size_t)a * DDIM_;
    float s = 0.0f;
    for (int d = 0; d < DDIM_; ++d) s += sq[d] * w[d];
    qp[b * ADIM_ + a] = s;
  }
}

__global__ __launch_bounds__(256) void zero_kernel(float* __restrict__ p, int n) {
  int i = blockIdx.x * 256 + threadIdx.x;
  if (i < n) p[i] = 0.0f;
}

// ---------------------------------------------------------------------------
// Fused GEMM + tanh + v-dot:
//   e[b][m] = sum_a v[a] * tanh( sum_d key[b][m][d]*Wk[a][d] + bk[a] + qp[b][a] )
// Workgroup: 8 waves, tile M=32 (klen rows) x N=512 (full a-dim), K chunked by 32.
// Wave w: rows (w>>2)*16 .. +16, cols (w&3)*128 .. +128  -> 8 WMMA acc tiles.
// Global->reg prefetch of the next K-chunk is overlapped with the WMMA section.
// ---------------------------------------------------------------------------
__global__ __launch_bounds__(256) void bahdanau_e_kernel(
    const float* __restrict__ key, const float* __restrict__ Wk,
    const float* __restrict__ bk,  const float* __restrict__ v,
    const float* __restrict__ qp,  float* __restrict__ e_out) {
  __shared__ __align__(16) _Float16 sWk[ADIM_][HPAD_];  // 40 KB (f16 chunk of Wk)
  __shared__ __align__(16) _Float16 sA[32][HPAD_];      // 2.5 KB (f16 chunk of key)
  __shared__ float combo[ADIM_];                        // bk[a] + qp[b][a]
  __shared__ float vv[ADIM_];
  __shared__ float se[32];                              // per-row e accumulator

  const int tid   = threadIdx.x;
  const int b     = blockIdx.y;
  const int mBase = blockIdx.x * 32;

  for (int n = tid; n < ADIM_; n += 256) {
    combo[n] = bk[n] + qp[b * ADIM_ + n];
    vv[n]    = v[n];
  }
  if (tid < 32) se[tid] = 0.0f;

  const int lane   = tid & 31;
  const int wave   = tid >> 5;
  const int laneHi = (lane >= 16) ? 1 : 0;
  const int l16    = lane & 15;
  const int mSub   = (wave >> 2) * 16;   // 0 or 16
  const int nGrp   = (wave & 3) * 128;   // 0,128,256,384

  v8f acc[8];
#pragma unroll
  for (int s = 0; s < 8; ++s) { v8f z = {}; acc[s] = z; }

  const float* keyBase = key + ((size_t)b * KLEN_ + mBase) * DDIM_;
  const int rA = tid >> 3;
  const int jA = (tid & 7) * 4;

  // Prefetch registers for the next K-chunk (hidden behind WMMA section).
  float4 fA;
  float4 fW[2][8];
  auto issue_loads = [&](int kc) {
    fA = *reinterpret_cast<const float4*>(keyBase + (size_t)rA * DDIM_ + kc + jA);
#pragma unroll
    for (int rr = 0; rr < 2; ++rr) {
      const float4* p =
          reinterpret_cast<const float4*>(Wk + (size_t)(tid + rr * 256) * DDIM_ + kc);
#pragma unroll
      for (int j = 0; j < 8; ++j) fW[rr][j] = p[j];
    }
  };

  issue_loads(0);

  for (int kc = 0; kc < DDIM_; kc += 32) {
    __syncthreads();
    // Convert + store the prefetched chunk to LDS.
    {
      _Float16* d = &sA[rA][jA];
      d[0] = (_Float16)fA.x; d[1] = (_Float16)fA.y;
      d[2] = (_Float16)fA.z; d[3] = (_Float16)fA.w;
    }
#pragma unroll
    for (int rr = 0; rr < 2; ++rr) {
      _Float16* d = &sWk[tid + rr * 256][0];
#pragma unroll
      for (int j = 0; j < 8; ++j) {
        float4 f = fW[rr][j];
        d[j * 4 + 0] = (_Float16)f.x; d[j * 4 + 1] = (_Float16)f.y;
        d[j * 4 + 2] = (_Float16)f.z; d[j * 4 + 3] = (_Float16)f.w;
      }
    }
    __syncthreads();

    // Prefetch next chunk (wraps to chunk 0 on the last iteration, harmless).
    issue_loads((kc + 32) & (DDIM_ - 1));

    // A fragment (16x32): lane<16 -> halves {0..7,16..23}, lane>=16 -> {8..15,24..31}
    const _Float16* ap = &sA[mSub + l16][laneHi * 8];
    v8h alo = *reinterpret_cast<const v8h*>(ap);
    v8h ahi = *reinterpret_cast<const v8h*>(ap + 16);
    v16h afrag = __builtin_shufflevector(alo, ahi,
        0, 1, 2, 3, 4, 5, 6, 7, 8, 9, 10, 11, 12, 13, 14, 15);

#pragma unroll
    for (int s = 0; s < 8; ++s) {
      const _Float16* bp = &sWk[nGrp + s * 16 + l16][laneHi * 8];
      v8h blo = *reinterpret_cast<const v8h*>(bp);
      v8h bhi = *reinterpret_cast<const v8h*>(bp + 16);
      v16h bfrag = __builtin_shufflevector(blo, bhi,
          0, 1, 2, 3, 4, 5, 6, 7, 8, 9, 10, 11, 12, 13, 14, 15);
      acc[s] = __builtin_amdgcn_wmma_f32_16x16x32_f16(
          false, afrag, false, bfrag, (short)0, acc[s], false, false);
    }
  }

  // Epilogue: t = tanh(c + bk + q) * v, reduce over a-dim into se[row].
  // C/D layout: VGPR r -> row r (lanes 0-15) / row r+8 (lanes 16-31), col = lane&15.
  float rowsum[8];
#pragma unroll
  for (int r = 0; r < 8; ++r) rowsum[r] = 0.0f;
#pragma unroll
  for (int s = 0; s < 8; ++s) {
    const int n = nGrp + s * 16 + l16;
    const float cb = combo[n];
    const float vn = vv[n];
#pragma unroll
    for (int r = 0; r < 8; ++r)
      rowsum[r] += TANH_(acc[s][r] + cb) * vn;
  }
#pragma unroll
  for (int r = 0; r < 8; ++r) {
    float x = rowsum[r];
    x += __shfl_xor(x, 1);
    x += __shfl_xor(x, 2);
    x += __shfl_xor(x, 4);
    x += __shfl_xor(x, 8);
    if (l16 == 0) atomicAdd(&se[mSub + laneHi * 8 + r], x);
  }
  __syncthreads();
  if (tid < 32) e_out[(size_t)b * KLEN_ + mBase + tid] = se[tid];
}

// ---------------------------------------------------------------------------
// softmax over klen with mask
// ---------------------------------------------------------------------------
__global__ __launch_bounds__(256) void softmax_kernel(
    const float* __restrict__ e, const int* __restrict__ mask,
    float* __restrict__ aw) {
  __shared__ float red[256];
  const int b = blockIdx.x;
  const int t = threadIdx.x;
  float vals[8];
  float mx = NEG_INF_;
#pragma unroll
  for (int i = 0; i < 8; ++i) {
    const int k = t + i * 256;
    float x = e[(size_t)b * KLEN_ + k];
    if (mask[(size_t)b * KLEN_ + k] == 0) x = NEG_INF_;
    vals[i] = x;
    mx = fmaxf(mx, x);
  }
  red[t] = mx; __syncthreads();
  for (int s = 128; s > 0; s >>= 1) {
    if (t < s) red[t] = fmaxf(red[t], red[t + s]);
    __syncthreads();
  }
  mx = red[0]; __syncthreads();
  float sum = 0.0f;
#pragma unroll
  for (int i = 0; i < 8; ++i) { vals[i] = __expf(vals[i] - mx); sum += vals[i]; }
  red[t] = sum; __syncthreads();
  for (int s = 128; s > 0; s >>= 1) {
    if (t < s) red[t] += red[t + s];
    __syncthreads();
  }
  const float inv = 1.0f / red[0];
#pragma unroll
  for (int i = 0; i < 8; ++i)
    aw[(size_t)b * KLEN_ + t + i * 256] = vals[i] * inv;
}

// ---------------------------------------------------------------------------
// cv[b][d] = sum_k aw[b][k] * value[b][k][d]; k split 8-way, atomic combine
// ---------------------------------------------------------------------------
__global__ __launch_bounds__(256) void cv_kernel(
    const float* __restrict__ aw, const float* __restrict__ value,
    float* __restrict__ cv) {
  const int b  = blockIdx.y;
  const int kc = blockIdx.x * 256;
  const int t  = threadIdx.x;
  const float* vb = value + ((size_t)b * KLEN_ + kc) * VDIM_;
  const float* ab = aw + (size_t)b * KLEN_ + kc;
  float a0 = 0.0f, a1 = 0.0f;
  for (int k = 0; k < 256; ++k) {
    const float w = ab[k];
    a0 += w * vb[(size_t)k * VDIM_ + t];
    a1 += w * vb[(size_t)k * VDIM_ + t + 256];
  }
  atomicAdd(&cv[b * VDIM_ + t], a0);
  atomicAdd(&cv[b * VDIM_ + t + 256], a1);
}

// ---------------------------------------------------------------------------
extern "C" void kernel_launch(void* const* d_in, const int* in_sizes, int n_in,
                              void* d_out, int out_size, void* d_ws, size_t ws_size,
                              hipStream_t stream) {
  (void)in_sizes; (void)n_in; (void)out_size; (void)ws_size;
  const float* key   = (const float*)d_in[0];
  const float* value = (const float*)d_in[1];
  const float* query = (const float*)d_in[2];
  const int*   mask  = (const int*)d_in[3];
  const float* Wk    = (const float*)d_in[4];
  const float* bk    = (const float*)d_in[5];
  const float* Wq    = (const float*)d_in[6];
  const float* v     = (const float*)d_in[7];

  float* cv = (float*)d_out;                 // [32,1,512]
  float* aw = cv + B_ * VDIM_;               // [32,1,1,2048]
  float* qp = (float*)d_ws;                  // [32,512]
  float* e  = qp + B_ * ADIM_;               // [32,2048]

  qproj_kernel<<<B_, 256, 0, stream>>>(query, Wq, qp);
  zero_kernel<<<(B_ * VDIM_ + 255) / 256, 256, 0, stream>>>(cv, B_ * VDIM_);
  bahdanau_e_kernel<<<dim3(KLEN_ / 32, B_), 256, 0, stream>>>(key, Wk, bk, v, qp, e);
  softmax_kernel<<<B_, 256, 0, stream>>>(e, mask, aw);
  cv_kernel<<<dim3(KLEN_ / 256, B_), 256, 0, stream>>>(aw, value, cv);
}
